// VQDecoder_62405874811771
// MI455X (gfx1250) — matrix operand order repaired and checked by reading, compiled, not run on hardware
//
#include <hip/hip_runtime.h>
#include <hip/hip_bf16.h>

// ---------------------------------------------------------------------------
// Swin block (B=32, H=W=64, C=256, WS=8, SS=4, heads=8, d=32, MLP=1024)
// One workgroup (256 thr = 8 wave32) per window (2048 windows).
// GEMMs via v_wmma_f32_16x16x32_f16; window input staged to LDS with
// global_load_async_to_lds_b128 (ASYNCcnt path).
// ---------------------------------------------------------------------------

typedef __attribute__((ext_vector_type(16))) _Float16 v16h;
typedef __attribute__((ext_vector_type(8)))  _Float16 v8h;
typedef __attribute__((ext_vector_type(8)))  float    v8f;

#define DIMC      256
#define WSZ       8
#define SSZ       4
#define NTOK      64
#define HEADS     8
#define HDIM      32
#define HID       1024

#define XH_STRIDE 264     // halves (or floats for xres); 16B-aligned rows
#define VT_STRIDE 72
#define PW_STRIDE 72
#define GH_STRIDE 1032

#define OFF_XH    0                       // 64 x 264 f16  (later: attn output Oh)
#define OFF_QH    33792                   // 64 x 264 f16  (later: LN2 output Hh)
#define OFF_XRES  67584                   // 64 x 264 f32  (shortcut / residual)
#define OFF_KH    135168                  // 64 x 264 f16
#define OFF_VT    168960                  // 256 x 72 f16  (V transposed)
#define OFF_PW    205824                  // 8 x 64 x 72 f16 (per-wave softmax P)
#define OFF_GH    135168                  // 64 x 1032 f16 (overlays KH/VT/PW)
#define SMEM_BYTES 279552

// ---- WMMA helpers ---------------------------------------------------------

__device__ __forceinline__ v8f wmma16(v16h a, v16h b, v8f c) {
  return __builtin_amdgcn_wmma_f32_16x16x32_f16(false, a, false, b, (short)0, c,
                                                false, false);
}

// A-matrix fragment (16x32, f16), row-major [M][K], `stride` halves per row.
// Per ISA 7.12.2: lanes 0-15 hold K {0..7,16..23}, lanes 16-31 {8..15,24..31}.
__device__ __forceinline__ v16h load_fragA(const _Float16* base, int stride,
                                           int rowBase, int kOff) {
  const int lane = threadIdx.x & 31;
  const int kb = (lane & 16) ? 8 : 0;
  const _Float16* p = base + (rowBase + (lane & 15)) * stride + kOff + kb;
  v8h lo = *(const v8h*)p;
  v8h hi = *(const v8h*)(p + 16);
  v16h r;
#pragma unroll
  for (int j = 0; j < 8; ++j) { r[j] = lo[j]; r[j + 8] = hi[j]; }
  return r;
}

// B-matrix fragment (32x16, f16) from a TRANSPOSED store [N][K]:
// lane holds column N = colBase+(lane&15); lanes 0-15 K {0..7,16..23}, 16-31 {8..15,24..31}.
__device__ __forceinline__ v16h load_fragB(const _Float16* base, int stride,
                                           int colBase, int kOff) {
  const int lane = threadIdx.x & 31;
  const _Float16* p = base + (colBase + (lane & 15)) * stride + kOff +
                      ((lane & 16) ? 8 : 0);
  v8h lo = *(const v8h*)p;
  v8h hi = *(const v8h*)(p + 16);
  v16h r;
#pragma unroll
  for (int j = 0; j < 8; ++j) { r[j] = lo[j]; r[j + 8] = hi[j]; }
  return r;
}

// ---- prep kernels ---------------------------------------------------------

// out[n*K + k] = (f16) in[k*N + n]   (transpose + downconvert)
__global__ void swin_w2h_t(const float* __restrict__ in, _Float16* __restrict__ out,
                           int K, int N) {
  int idx = blockIdx.x * blockDim.x + threadIdx.x;
  if (idx >= K * N) return;
  int n = idx / K, k = idx - n * K;
  out[(size_t)n * K + k] = (_Float16)in[(size_t)k * N + n];
}

// biasFull[h][n][m] = rpb[rpi(n,m)*8 + h]
__global__ void swin_bias_expand(const float* __restrict__ rpb,
                                 float* __restrict__ biasFull) {
  int idx = blockIdx.x * blockDim.x + threadIdx.x;
  if (idx >= HEADS * NTOK * NTOK) return;
  int m = idx & 63, n = (idx >> 6) & 63, h = idx >> 12;
  int rn = n >> 3, cn = n & 7, rm = m >> 3, cm = m & 7;
  int rpi = (rn - rm + 7) * 15 + (cn - cm + 7);
  biasFull[idx] = rpb[rpi * HEADS + h];
}

// ---- main fused kernel ----------------------------------------------------

__global__ __launch_bounds__(256, 1)
void swin_block_fused(const float* __restrict__ x,
                      const float* __restrict__ n1g, const float* __restrict__ n1b,
                      const float* __restrict__ qkv_b,
                      const float* __restrict__ proj_b,
                      const float* __restrict__ n2g, const float* __restrict__ n2b,
                      const float* __restrict__ fc1_b, const float* __restrict__ fc2_b,
                      const _Float16* __restrict__ qkv_wt,
                      const _Float16* __restrict__ proj_wt,
                      const _Float16* __restrict__ fc1_wt,
                      const _Float16* __restrict__ fc2_wt,
                      const float* __restrict__ biasF,
                      float* __restrict__ out) {
  extern __shared__ __align__(16) unsigned char smem[];
  _Float16* Xh   = (_Float16*)(smem + OFF_XH);    // LN1 out, then attn-out Oh
  _Float16* Qh   = (_Float16*)(smem + OFF_QH);    // Q (scaled), then LN2 out Hh
  float*    xres = (float*)   (smem + OFF_XRES);  // shortcut / residual, f32
  _Float16* Kh   = (_Float16*)(smem + OFF_KH);
  _Float16* Vt   = (_Float16*)(smem + OFF_VT);    // V transposed [d][m]
  _Float16* Pw   = (_Float16*)(smem + OFF_PW);    // per-wave softmax probs
  _Float16* Gh   = (_Float16*)(smem + OFF_GH);    // MLP hidden [64][1032]

  const int tid  = threadIdx.x;
  const int lane = tid & 31;
  const int wvu  = __builtin_amdgcn_readfirstlane(tid >> 5);  // scalar wave id
  const int l15  = lane & 15;
  const int hh8  = (lane & 16) ? 8 : 0;

  const int blk  = blockIdx.x;
  const int bb   = blk >> 6;          // batch
  const int wIdx = blk & 63;          // window index (shifted frame)
  const int wi   = wIdx >> 3, wj = wIdx & 7;

  // token t (in shifted window) -> flat element offset in original x / out
  auto tokBase = [&](int t) -> size_t {
    int hs = wi * WSZ + (t >> 3), ws_ = wj * WSZ + (t & 7);
    int hg = (hs + SSZ) & 63, wg = (ws_ + SSZ) & 63;
    return ((size_t)bb * 4096 + (size_t)(hg * 64 + wg)) * DIMC;
  };
  // shift-mask region id of token t
  auto region = [&](int t) -> int {
    int hs = wi * WSZ + (t >> 3), ws_ = wj * WSZ + (t & 7);
    int a = (hs < 56) ? 0 : (hs < 60 ? 1 : 2);
    int c = (ws_ < 56) ? 0 : (ws_ < 60 ? 1 : 2);
    return a * 3 + c;
  };

  // ---------------- Phase A0: async-copy window of x into xres (LDS) -----
  // 64 tokens x 256 f32 = 4096 x 16B chunks; ASYNCcnt-tracked direct
  // global->LDS moves (no VGPR round trip). No static LDS in this kernel,
  // so dynamic-LDS offsets are the hardware LDS addresses.
  {
    const uint64_t xbase = (uint64_t)(uintptr_t)x;
#pragma unroll
    for (int i = 0; i < 16; ++i) {
      int c = i * 256 + tid;            // chunk id 0..4095
      int t = c >> 6, p = c & 63;       // token, 16B chunk within token row
      uint64_t g = xbase + ((uint64_t)tokBase(t) + (uint64_t)(p * 4)) * 4ull;
      uint32_t l = (uint32_t)(OFF_XRES + t * (XH_STRIDE * 4) + p * 16);
      asm volatile("global_load_async_to_lds_b128 %0, %1, off"
                   :: "v"(l), "v"(g) : "memory");
    }
    asm volatile("s_wait_asynccnt 0x0" ::: "memory");
  }
  __syncthreads();

  // ---------------- Phase A1: LN1 ---------------------------------------
  {
    const int t = wvu * 8 + (lane >> 2);     // token, 4 lanes per token
    const int cb = (lane & 3) * 64;
    float s = 0.f, s2 = 0.f;
#pragma unroll 8
    for (int i = 0; i < 64; ++i) {
      float v = xres[t * XH_STRIDE + cb + i];
      s += v; s2 += v * v;
    }
    s  += __shfl_xor(s, 1);  s  += __shfl_xor(s, 2);
    s2 += __shfl_xor(s2, 1); s2 += __shfl_xor(s2, 2);
    float mu = s * (1.f / 256.f);
    float rs = rsqrtf(s2 * (1.f / 256.f) - mu * mu + 1e-5f);
#pragma unroll 8
    for (int i = 0; i < 64; ++i) {
      int ch = cb + i;
      float y = (xres[t * XH_STRIDE + ch] - mu) * rs * n1g[ch] + n1b[ch];
      Xh[t * XH_STRIDE + ch] = (_Float16)y;
    }
  }
  __syncthreads();

  // ---------------- Phase B: QKV = Xh @ Wqkv + b ------------------------
  {
#pragma unroll 1
    for (int it = 0; it < 24; ++it) {
      int tix = wvu + 8 * it;                // 0..191 = 48 N-tiles x 4 M-tiles
      int mr = (tix & 3) * 16, nc = (tix >> 2) * 16;
      v8f acc = {};
#pragma unroll
      for (int k8 = 0; k8 < 8; ++k8) {
        if (k8 < 7)
          __builtin_prefetch(qkv_wt + (nc + l15) * DIMC + (k8 + 1) * 32, 0, 1);
        acc = wmma16(load_fragA(Xh, XH_STRIDE, mr, k8 * 32),
                     load_fragB(qkv_wt, DIMC, nc, k8 * 32), acc);
      }
      const int colG = nc + l15;
      const float bias = qkv_b[colG];
      const int rb = mr + hh8;
      if (nc < 256) {                        // Q (fold in 1/sqrt(d)); scalar branch
        const float sc = 0.17677669529663687f;
#pragma unroll
        for (int v = 0; v < 8; ++v)
          Qh[(rb + v) * XH_STRIDE + colG] = (_Float16)((acc[v] + bias) * sc);
      } else if (nc < 512) {                 // K
        int cc = colG - 256;
#pragma unroll
        for (int v = 0; v < 8; ++v)
          Kh[(rb + v) * XH_STRIDE + cc] = (_Float16)(acc[v] + bias);
      } else {                               // V, stored transposed [d][m]
        int cc = colG - 512;
#pragma unroll
        for (int v = 0; v < 8; ++v)
          Vt[cc * VT_STRIDE + (rb + v)] = (_Float16)(acc[v] + bias);
      }
    }
  }
  __syncthreads();

  // ---------------- Phase C: attention, one wave per head ---------------
  {
    const int h = wvu;
    v16h aq[4];
#pragma unroll
    for (int i = 0; i < 4; ++i) aq[i] = load_fragA(Qh, XH_STRIDE, i * 16, h * HDIM);

    v8f s[4][4];
#pragma unroll
    for (int j = 0; j < 4; ++j) {
      v16h bk = load_fragB(Kh, XH_STRIDE, j * 16, h * HDIM);
#pragma unroll
      for (int i = 0; i < 4; ++i) { v8f z = {}; s[i][j] = wmma16(aq[i], bk, z); }
    }

    // rel-pos bias + shift mask
#pragma unroll
    for (int j = 0; j < 4; ++j) {
      const int mcol = j * 16 + l15;
      const int regM = region(mcol);
#pragma unroll
      for (int i = 0; i < 4; ++i) {
#pragma unroll
        for (int v = 0; v < 8; ++v) {
          int nrow = i * 16 + hh8 + v;
          float add = biasF[(h * NTOK + nrow) * NTOK + mcol];
          if (region(nrow) != regM) add -= 100.f;
          s[i][j][v] += add;
        }
      }
    }

    // exact softmax (row = n, across 4 col-tiles + 16 lanes of same half)
    _Float16* Pme = Pw + wvu * NTOK * PW_STRIDE;
#pragma unroll
    for (int i = 0; i < 4; ++i) {
#pragma unroll
      for (int v = 0; v < 8; ++v) {
        float mx = fmaxf(fmaxf(s[i][0][v], s[i][1][v]),
                         fmaxf(s[i][2][v], s[i][3][v]));
        mx = fmaxf(mx, __shfl_xor(mx, 1));
        mx = fmaxf(mx, __shfl_xor(mx, 2));
        mx = fmaxf(mx, __shfl_xor(mx, 4));
        mx = fmaxf(mx, __shfl_xor(mx, 8));
        float sm = 0.f;
#pragma unroll
        for (int j = 0; j < 4; ++j) {
          s[i][j][v] = __expf(s[i][j][v] - mx);
          sm += s[i][j][v];
        }
        sm += __shfl_xor(sm, 1); sm += __shfl_xor(sm, 2);
        sm += __shfl_xor(sm, 4); sm += __shfl_xor(sm, 8);
        float inv = 1.f / sm;
        int nrow = i * 16 + hh8 + v;
#pragma unroll
        for (int j = 0; j < 4; ++j)
          Pme[nrow * PW_STRIDE + j * 16 + l15] = (_Float16)(s[i][j][v] * inv);
      }
    }

    // O = P @ V_h  -> Oh (reuses Xh region), f16
#pragma unroll
    for (int i = 0; i < 4; ++i) {
      v16h ap0 = load_fragA(Pme, PW_STRIDE, i * 16, 0);
      v16h ap1 = load_fragA(Pme, PW_STRIDE, i * 16, 32);
#pragma unroll
      for (int di = 0; di < 2; ++di) {
        v8f o = {};
        o = wmma16(ap0, load_fragB(Vt, VT_STRIDE, h * HDIM + di * 16, 0), o);
        o = wmma16(ap1, load_fragB(Vt, VT_STRIDE, h * HDIM + di * 16, 32), o);
        int colG = h * HDIM + di * 16 + l15;
        int rb = i * 16 + hh8;
#pragma unroll
        for (int v = 0; v < 8; ++v)
          Xh[(rb + v) * XH_STRIDE + colG] = (_Float16)o[v];
      }
    }
  }
  __syncthreads();

  // ---------------- Phase D: proj + residual into xres -------------------
  {
#pragma unroll 1
    for (int it = 0; it < 8; ++it) {
      int tix = wvu + 8 * it;                // 0..63
      int mr = (tix & 3) * 16, nc = (tix >> 2) * 16;
      v8f acc = {};
#pragma unroll
      for (int k8 = 0; k8 < 8; ++k8)
        acc = wmma16(load_fragA(Xh, XH_STRIDE, mr, k8 * 32),
                     load_fragB(proj_wt, DIMC, nc, k8 * 32), acc);
      int colG = nc + l15;
      float bias = proj_b[colG];
      int rb = mr + hh8;
#pragma unroll
      for (int v = 0; v < 8; ++v)
        xres[(rb + v) * XH_STRIDE + colG] += acc[v] + bias;
    }
  }
  __syncthreads();

  // ---------------- Phase E: LN2 -> Hh (reuses Qh) -----------------------
  {
    const int t = wvu * 8 + (lane >> 2);
    const int cb = (lane & 3) * 64;
    float s = 0.f, s2 = 0.f;
#pragma unroll 8
    for (int i = 0; i < 64; ++i) {
      float v = xres[t * XH_STRIDE + cb + i];
      s += v; s2 += v * v;
    }
    s  += __shfl_xor(s, 1);  s  += __shfl_xor(s, 2);
    s2 += __shfl_xor(s2, 1); s2 += __shfl_xor(s2, 2);
    float mu = s * (1.f / 256.f);
    float rs = rsqrtf(s2 * (1.f / 256.f) - mu * mu + 1e-5f);
#pragma unroll 8
    for (int i = 0; i < 64; ++i) {
      int ch = cb + i;
      float y = (xres[t * XH_STRIDE + ch] - mu) * rs * n2g[ch] + n2b[ch];
      Qh[t * XH_STRIDE + ch] = (_Float16)y;
    }
  }
  __syncthreads();

  // ---------------- Phase F: fc1 + exact GELU -> Gh ----------------------
  {
#pragma unroll 1
    for (int it = 0; it < 32; ++it) {
      int tix = wvu + 8 * it;                // 0..255 = 64 N-tiles x 4 M-tiles
      int mr = (tix & 3) * 16, nc = (tix >> 2) * 16;
      v8f acc = {};
#pragma unroll
      for (int k8 = 0; k8 < 8; ++k8) {
        if (k8 < 7)
          __builtin_prefetch(fc1_wt + (nc + l15) * DIMC + (k8 + 1) * 32, 0, 1);
        acc = wmma16(load_fragA(Qh, XH_STRIDE, mr, k8 * 32),
                     load_fragB(fc1_wt, DIMC, nc, k8 * 32), acc);
      }
      int colG = nc + l15;
      float bias = fc1_b[colG];
      int rb = mr + hh8;
#pragma unroll
      for (int v = 0; v < 8; ++v) {
        float y = acc[v] + bias;
        float g = 0.5f * y * (1.f + erff(y * 0.70710678118654752f));
        Gh[(rb + v) * GH_STRIDE + colG] = (_Float16)g;
      }
    }
  }
  __syncthreads();

  // ---------------- Phase G: fc2 + residual -> out (inverse shift) -------
  {
#pragma unroll 1
    for (int it = 0; it < 8; ++it) {
      int tix = wvu + 8 * it;
      int mr = (tix & 3) * 16, nc = (tix >> 2) * 16;
      v8f acc = {};
#pragma unroll 4
      for (int k8 = 0; k8 < 32; ++k8)
        acc = wmma16(load_fragA(Gh, GH_STRIDE, mr, k8 * 32),
                     load_fragB(fc2_wt, HID, nc, k8 * 32), acc);
      int colG = nc + l15;
      float bias = fc2_b[colG];
      int rb = mr + hh8;
#pragma unroll
      for (int v = 0; v < 8; ++v) {
        int t = rb + v;
        float r = xres[t * XH_STRIDE + colG] + acc[v] + bias;
        out[tokBase(t) + colG] = r;
      }
    }
  }
}

// ---------------------------------------------------------------------------

extern "C" void kernel_launch(void* const* d_in, const int* in_sizes, int n_in,
                              void* d_out, int out_size, void* d_ws, size_t ws_size,
                              hipStream_t stream) {
  (void)in_sizes; (void)n_in; (void)out_size; (void)ws_size;
  const float* x      = (const float*)d_in[0];
  const float* n1g    = (const float*)d_in[1];
  const float* n1b    = (const float*)d_in[2];
  const float* qkv_w  = (const float*)d_in[3];
  const float* qkv_b  = (const float*)d_in[4];
  const float* proj_w = (const float*)d_in[5];
  const float* proj_b = (const float*)d_in[6];
  const float* rpb    = (const float*)d_in[7];
  const float* n2g    = (const float*)d_in[8];
  const float* n2b    = (const float*)d_in[9];
  const float* fc1_w  = (const float*)d_in[10];
  const float* fc1_b  = (const float*)d_in[11];
  const float* fc2_w  = (const float*)d_in[12];
  const float* fc2_b  = (const float*)d_in[13];

  char* ws = (char*)d_ws;
  _Float16* qkv_wt  = (_Float16*)(ws);                       // [768][256] f16
  _Float16* proj_wt = (_Float16*)(ws + 393216);              // [256][256]
  _Float16* fc1_wt  = (_Float16*)(ws + 393216 + 131072);     // [1024][256]
  _Float16* fc2_wt  = (_Float16*)(ws + 393216 + 131072 + 524288);          // [256][1024]
  float*    biasF   = (float*)   (ws + 393216 + 131072 + 524288 + 524288); // [8][64][64]

  swin_w2h_t<<<(256 * 768 + 255) / 256, 256, 0, stream>>>(qkv_w, qkv_wt, 256, 768);
  swin_w2h_t<<<(256 * 256 + 255) / 256, 256, 0, stream>>>(proj_w, proj_wt, 256, 256);
  swin_w2h_t<<<(256 * 1024 + 255) / 256, 256, 0, stream>>>(fc1_w, fc1_wt, 256, 1024);
  swin_w2h_t<<<(1024 * 256 + 255) / 256, 256, 0, stream>>>(fc2_w, fc2_wt, 1024, 256);
  swin_bias_expand<<<(HEADS * NTOK * NTOK + 255) / 256, 256, 0, stream>>>(rpb, biasF);

  (void)hipFuncSetAttribute(reinterpret_cast<const void*>(swin_block_fused),
                            hipFuncAttributeMaxDynamicSharedMemorySize, SMEM_BYTES);
  swin_block_fused<<<2048, 256, SMEM_BYTES, stream>>>(
      x, n1g, n1b, qkv_b, proj_b, n2g, n2b, fc1_b, fc2_b,
      qkv_wt, proj_wt, fc1_wt, fc2_wt, biasF, (float*)d_out);
}
